// DeepInterestNetwork_31628139167809
// MI455X (gfx1250) — matrix operand order repaired and checked by reading, not compiled, not run on hardware
//
#include <hip/hip_runtime.h>

#define ITEM_NUM 100000
#define EDIM     96
#define BF       69     // F
#define BATCH    8192
#define S_DICE   0.99999999950f    // 1/sqrt(1+1e-9)
#define INV_BN5  0.99999500003750f // 1/sqrt(1+1e-5)

typedef __attribute__((ext_vector_type(8)))  float  v8f;
typedef __attribute__((ext_vector_type(16))) __bf16 v16bf;

union Frag { v16bf v; unsigned u[8]; uint4 q[2]; };

__device__ __forceinline__ unsigned packbf(float lo, float hi) {
    union { __bf16 b[2]; unsigned u; } cv;
    cv.b[0] = (__bf16)lo;           // native v_cvt, RNE
    cv.b[1] = (__bf16)hi;
    return cv.u;
}
__device__ __forceinline__ float dicef(float x, float alpha) {
    float p = 1.0f / (1.0f + __expf(-x * S_DICE));
    return (alpha * (1.0f - p) + p) * x;
}
__device__ __forceinline__ int kpair_lo(int i, int kh) {   // A-frag K index of VGPR i, low half
    return 2 * i + 16 * kh;
}

// ---------------- Kernel 0: pack weights to bf16 B-fragments; fold W2@W3 ----------------
// B-frag layout (this code's contiguous K order, 16-bit B 32x16): lane l -> n = nt*16 + (l&15); khalf=l>>4;
// dword i holds K = kt*32 + 16*khalf + 2*i (lo) and +1 (hi).
// pk dword index = (fragIdx*32 + lane)*8 + i, fragIdx = kt*NT + nt.
#define PKD_W1  12288    // 12*4*256
#define PKD_WF1 109824   // 33*13*256
#define PKD_ALL 131072   // + 7*5*256 (Wf2)

__global__ __launch_bounds__(256) void din_prep(
    const float* __restrict__ W1, const float* __restrict__ W2, const float* __restrict__ W3,
    const float* __restrict__ b2, const float* __restrict__ b3,
    const float* __restrict__ Wf1, const float* __restrict__ Wf2,
    unsigned* __restrict__ pk, float* __restrict__ w23, float* __restrict__ c23)
{
    int idx = blockIdx.x * 256 + threadIdx.x;
    if (idx < PKD_ALL) {
        const float* W; int K, N, NT, rel;
        if (idx < PKD_W1)                 { W = W1;  K = 384;  N = 64;  NT = 4;  rel = idx; }
        else if (idx < PKD_W1 + PKD_WF1)  { W = Wf1; K = 1056; N = 200; NT = 13; rel = idx - PKD_W1; }
        else                              { W = Wf2; K = 200;  N = 80;  NT = 5;  rel = idx - (PKD_W1 + PKD_WF1); }
        int frag = rel >> 8, within = rel & 255;
        int lane = within >> 3, i = within & 7;
        int kt = frag / NT, nt = frag - kt * NT;
        int k = kt * 32 + 16 * (lane >> 4) + 2 * i;
        int n = nt * 16 + (lane & 15);
        float lo = (k     < K && n < N) ? W[(size_t)k       * N + n] : 0.0f;
        float hi = (k + 1 < K && n < N) ? W[(size_t)(k + 1) * N + n] : 0.0f;
        pk[idx] = packbf(lo, hi);
    } else if (idx < PKD_ALL + 64) {
        int j = idx - PKD_ALL;
        float s = 0.0f;
        #pragma unroll
        for (int t = 0; t < 16; ++t) s += W2[j * 16 + t] * W3[t];
        w23[j] = s;
        if (j == 0) {
            float c = b3[0];
            #pragma unroll
            for (int t = 0; t < 16; ++t) c += b2[t] * W3[t];
            c23[0] = c;
        }
    }
}

// ---------------- Kernel 1: attention MLP (WMMA) + masked group pooling ----------------
__global__ __launch_bounds__(256) void din_attention_pool(
    const int* __restrict__ bu, const int* __restrict__ bl, const float* __restrict__ emb,
    const float* __restrict__ b1, const float* __restrict__ a1,
    const unsigned* __restrict__ pkW1, const float* __restrict__ w23, const float* __restrict__ c23p,
    __bf16* __restrict__ xout)
{
    __shared__ float pooled[10 * EDIM];
    __shared__ float att_s[80];
    __shared__ int   user_sh[80];
    __shared__ int   label_sh;

    int b = blockIdx.x;
    int tid = threadIdx.x, lane = tid & 31, wv = tid >> 5;

    for (int i = tid; i < 10 * EDIM; i += 256) pooled[i] = 0.0f;
    if (tid < 80) user_sh[tid] = (tid < BF) ? bu[b * BF + tid] : ITEM_NUM;
    if (tid == 0) label_sh = bl[b];
    __syncthreads();

    const float* qrow = emb + (size_t)label_sh * EDIM;

    if (wv < 5) {                                   // 5 waves cover 80 >= 69 rows
        int m = lane & 15, kh = lane >> 4;
        int f = wv * 16 + m;
        const float* urow = emb + (size_t)user_sh[f] * EDIM;
        v8f c[4] = {};
        #pragma unroll
        for (int kt = 0; kt < 12; ++kt) {
            int seg = kt / 3, ebase = (kt % 3) * 32;
            Frag a;
            #pragma unroll
            for (int i = 0; i < 8; ++i) {
                int e = ebase + kpair_lo(i, kh);
                float lo, hi;
                if (seg == 0)      { lo = qrow[e];            hi = qrow[e + 1]; }
                else if (seg == 1) { lo = urow[e];            hi = urow[e + 1]; }
                else if (seg == 2) { lo = qrow[e] - urow[e];  hi = qrow[e + 1] - urow[e + 1]; }
                else               { lo = qrow[e] * urow[e];  hi = qrow[e + 1] * urow[e + 1]; }
                a.v[2 * i]     = (__bf16)lo;      // native cvt, RNE
                a.v[2 * i + 1] = (__bf16)hi;
            }
            #pragma unroll
            for (int nt = 0; nt < 4; ++nt) {
                const uint4* bp = (const uint4*)pkW1 + ((size_t)((kt * 4 + nt) * 32 + lane)) * 2;
                Frag bb; bb.q[0] = bp[0]; bb.q[1] = bp[1];
                c[nt] = __builtin_amdgcn_wmma_f32_16x16x32_bf16(
                            false, a.v, false, bb.v, (short)0, c[nt], false, false);
            }
        }
        // epilogue: bias + dice + fused (W2@W3) dot, cross-lane reduce
        int n0 = lane & 15;
        float pr[8] = {0, 0, 0, 0, 0, 0, 0, 0};
        #pragma unroll
        for (int nt = 0; nt < 4; ++nt) {
            int n = nt * 16 + n0;
            float bias = b1[n], al = a1[n], wv3 = w23[n];
            #pragma unroll
            for (int r = 0; r < 8; ++r) {
                float x = c[nt][r] + bias;
                pr[r] += dicef(x, al) * wv3;
            }
        }
        #pragma unroll
        for (int msk = 8; msk; msk >>= 1) {
            #pragma unroll
            for (int r = 0; r < 8; ++r) pr[r] += __shfl_xor(pr[r], msk, 32);
        }
        float cc = c23p[0];
        if ((lane & 15) == 0) {
            #pragma unroll
            for (int r = 0; r < 8; ++r) att_s[wv * 16 + r + 8 * kh] = pr[r] + cc;
        }
    }
    __syncthreads();

    // masked group-sum pooling via LDS float atomics (ds_add_f32)
    for (int idx = tid; idx < BF * EDIM; idx += 256) {
        int f = idx / EDIM, e = idx - f * EDIM;
        int usr = user_sh[f];
        if (usr < ITEM_NUM) {
            int g = (f < 20) ? 0 : (f < 40) ? 1 : (f < 50) ? 2 : (f < 60) ? 3 :
                    (f < 62) ? 4 : (f < 64) ? 5 : (f < 66) ? 6 : (f < 67) ? 7 :
                    (f < 68) ? 8 : 9;
            atomicAdd(&pooled[g * EDIM + e], emb[(size_t)usr * EDIM + e] * att_s[f]);
        }
    }
    __syncthreads();

    // x = [pooled(960) | ql(96)] as bf16
    __bf16* xr = xout + (size_t)b * 1056;
    for (int idx = tid; idx < 1056; idx += 256) {
        float v = (idx < 960) ? pooled[idx] : qrow[idx - 960];
        xr[idx] = (__bf16)v;
    }
}

// ---------------- Kernel 2: x(8192x1056) @ Wf1 -> BN -> dice -> h (bf16, stride 224) ----------------
__global__ __launch_bounds__(256) void din_fc1(
    const unsigned* __restrict__ xin, const unsigned* __restrict__ pkWf1,
    const float* __restrict__ bf1, const float* __restrict__ af1,
    __bf16* __restrict__ hout)
{
    int lane = threadIdx.x & 31, wv = threadIdx.x >> 5;
    int mt = blockIdx.x * 8 + wv;
    int m = lane & 15, kh = lane >> 4, n0 = lane & 15;
    const unsigned* xrow = xin + (size_t)(mt * 16 + m) * (1056 / 2);
    v8f c[13] = {};
    #pragma unroll 1
    for (int kt = 0; kt < 33; ++kt) {
        Frag a;
        #pragma unroll
        for (int i = 0; i < 8; ++i)
            a.u[i] = xrow[(kt * 32 + kpair_lo(i, kh)) >> 1];
        if (kt + 1 < 33)
            __builtin_prefetch((const void*)((const uint4*)pkWf1 +
                               ((size_t)(((kt + 1) * 13) * 32 + lane)) * 2), 0, 0);
        #pragma unroll
        for (int nt = 0; nt < 13; ++nt) {
            const uint4* bp = (const uint4*)pkWf1 + ((size_t)((kt * 13 + nt) * 32 + lane)) * 2;
            Frag bb; bb.q[0] = bp[0]; bb.q[1] = bp[1];
            c[nt] = __builtin_amdgcn_wmma_f32_16x16x32_bf16(
                        false, a.v, false, bb.v, (short)0, c[nt], false, false);
        }
    }
    #pragma unroll
    for (int nt = 0; nt < 13; ++nt) {
        int n = nt * 16 + n0;
        float bias = (n < 200) ? bf1[n] : 0.0f;
        float al   = (n < 200) ? af1[n] : 0.0f;
        #pragma unroll
        for (int r = 0; r < 8; ++r) {
            int mrow = mt * 16 + r + 8 * kh;
            float v = (c[nt][r] + bias) * INV_BN5;
            float h = dicef(v, al);
            hout[(size_t)mrow * 224 + n] = (n < 200) ? (__bf16)h : (__bf16)0.0f;
        }
    }
    #pragma unroll
    for (int r = 0; r < 8; ++r) {   // zero K-pad cols 208..223
        int mrow = mt * 16 + r + 8 * kh;
        hout[(size_t)mrow * 224 + 208 + n0] = (__bf16)0.0f;
    }
}

// ---------------- Kernel 3: h @ Wf2 -> BN -> dice -> fused @Wf3 + bf3 -> out ----------------
__global__ __launch_bounds__(256) void din_fc23(
    const unsigned* __restrict__ hin, const unsigned* __restrict__ pkWf2,
    const float* __restrict__ bf2, const float* __restrict__ af2,
    const float* __restrict__ Wf3, const float* __restrict__ bf3, float* __restrict__ out)
{
    int lane = threadIdx.x & 31, wv = threadIdx.x >> 5;
    int mt = blockIdx.x * 8 + wv;
    int m = lane & 15, kh = lane >> 4, n0 = lane & 15;
    const unsigned* hrow = hin + (size_t)(mt * 16 + m) * (224 / 2);
    v8f c[5] = {};
    #pragma unroll 1
    for (int kt = 0; kt < 7; ++kt) {
        Frag a;
        #pragma unroll
        for (int i = 0; i < 8; ++i)
            a.u[i] = hrow[(kt * 32 + kpair_lo(i, kh)) >> 1];
        #pragma unroll
        for (int nt = 0; nt < 5; ++nt) {
            const uint4* bp = (const uint4*)pkWf2 + ((size_t)((kt * 5 + nt) * 32 + lane)) * 2;
            Frag bb; bb.q[0] = bp[0]; bb.q[1] = bp[1];
            c[nt] = __builtin_amdgcn_wmma_f32_16x16x32_bf16(
                        false, a.v, false, bb.v, (short)0, c[nt], false, false);
        }
    }
    float pr[8] = {0, 0, 0, 0, 0, 0, 0, 0};
    #pragma unroll
    for (int nt = 0; nt < 5; ++nt) {
        int n = nt * 16 + n0;
        float bias = bf2[n], al = af2[n], w3 = Wf3[n];
        #pragma unroll
        for (int r = 0; r < 8; ++r) {
            float v = (c[nt][r] + bias) * INV_BN5;
            pr[r] += dicef(v, al) * w3;
        }
    }
    #pragma unroll
    for (int msk = 8; msk; msk >>= 1) {
        #pragma unroll
        for (int r = 0; r < 8; ++r) pr[r] += __shfl_xor(pr[r], msk, 32);
    }
    float bb3 = bf3[0];
    if ((lane & 15) == 0) {
        #pragma unroll
        for (int r = 0; r < 8; ++r) out[mt * 16 + r + 8 * kh] = pr[r] + bb3;
    }
}

// ---------------- Host launcher ----------------
extern "C" void kernel_launch(void* const* d_in, const int* in_sizes, int n_in,
                              void* d_out, int out_size, void* d_ws, size_t ws_size,
                              hipStream_t stream) {
    (void)in_sizes; (void)n_in; (void)out_size; (void)ws_size;
    const int*   bu  = (const int*)d_in[0];
    const int*   bl  = (const int*)d_in[1];
    const float* emb = (const float*)d_in[2];
    const float* W1  = (const float*)d_in[3];
    const float* b1  = (const float*)d_in[4];
    const float* a1  = (const float*)d_in[5];
    const float* W2  = (const float*)d_in[6];
    const float* b2  = (const float*)d_in[7];
    const float* W3  = (const float*)d_in[8];
    const float* b3  = (const float*)d_in[9];
    const float* Wf1 = (const float*)d_in[10];
    const float* bf1 = (const float*)d_in[11];
    const float* af1 = (const float*)d_in[12];
    const float* Wf2 = (const float*)d_in[13];
    const float* bf2 = (const float*)d_in[14];
    const float* af2 = (const float*)d_in[15];
    const float* Wf3 = (const float*)d_in[16];
    const float* bf3 = (const float*)d_in[17];
    // d_in[18] = window (folded into hardcoded group table)

    char* ws = (char*)d_ws;
    unsigned* pk  = (unsigned*)ws;                      // 131072 u32 = 512 KB
    float*    w23 = (float*)(ws + 524288);              // 64 f32
    float*    c23 = (float*)(ws + 524544);              // 1 f32
    __bf16*   xb  = (__bf16*)(ws + 525312);             // 8192*1056 bf16 = 16.5 MB
    __bf16*   hb  = (__bf16*)(ws + 525312 + (size_t)BATCH * 1056 * 2); // 8192*224 bf16

    din_prep<<<513, 256, 0, stream>>>(W1, W2, W3, b2, b3, Wf1, Wf2, pk, w23, c23);
    din_attention_pool<<<BATCH, 256, 0, stream>>>(bu, bl, emb, b1, a1, pk, w23, c23, xb);
    din_fc1<<<64, 256, 0, stream>>>((const unsigned*)xb, pk + PKD_W1, bf1, af1, hb);
    din_fc23<<<64, 256, 0, stream>>>((const unsigned*)hb, pk + PKD_W1 + PKD_WF1,
                                     bf2, af2, Wf3, bf3, (float*)d_out);
}
